// GatedEdgeEncoder_9294309228899
// MI455X (gfx1250) — matrix operand order repaired and checked by reading, not compiled, not run on hardware
//
#include <hip/hip_runtime.h>
#include <hip/hip_bf16.h>
#include <stdint.h>

typedef __attribute__((ext_vector_type(16))) __bf16 v16bf;
typedef __attribute__((ext_vector_type(8)))  float  v8f;

#define NODE_DIM 128
#define EDGE_IN  31
#define EMB_DIM  128
#define GIN      287            // 2*128 + 31
#define GINP     288            // padded K for gate MLP (9 * 32)
#define HID      128
#define TILE_E   16
#define WAVES_PER_BLOCK 8
#define BLOCK_THREADS (WAVES_PER_BLOCK * 32)

__device__ __forceinline__ unsigned short f2bf(float f) {
  unsigned u = __float_as_uint(f);
  unsigned r = u + 0x7FFFu + ((u >> 16) & 1u);   // round-to-nearest-even
  return (unsigned short)(r >> 16);
}

__device__ __forceinline__ float geluf(float v) {
  float t = 0.7978845608028654f * (v + 0.044715f * v * v * v);
  return 0.5f * v * (1.0f + tanhf(t));
}

union Frag { uint4 q[2]; v16bf v; };

__global__ __launch_bounds__(BLOCK_THREADS)
void gated_edge_kernel(const float* __restrict__ x,
                       const float* __restrict__ edge_attr,
                       const int*   __restrict__ edge_index,
                       const float* __restrict__ Wp,
                       const float* __restrict__ bp,
                       const float* __restrict__ W1,
                       const float* __restrict__ b1,
                       const float* __restrict__ W2,
                       const float* __restrict__ b2,
                       float* __restrict__ out,
                       int E)
{
  // Weights transposed to [N][K] bf16 so a B-fragment is 32 contiguous bytes per lane.
  __shared__ __align__(16) unsigned short sWt1[HID * GINP];       // 73728 B
  __shared__ __align__(16) unsigned short sWtP[EMB_DIM * 32];     //  8192 B
  __shared__ __align__(16) unsigned short sGate[WAVES_PER_BLOCK * TILE_E * GINP]; // 73728 B
  __shared__ float sB1[HID];
  __shared__ float sBP[EMB_DIM];
  __shared__ float sW2[HID];

  const int tid = threadIdx.x;

  // ---- stage weights: f32 -> bf16, transpose to [N][K] ----
  for (int i = tid; i < GIN * HID; i += BLOCK_THREADS) {
    int k = i >> 7;          // row of W1 (K index), coalesced read over n
    int n = i & 127;
    sWt1[n * GINP + k] = f2bf(W1[i]);
  }
  for (int i = tid; i < EDGE_IN * EMB_DIM; i += BLOCK_THREADS) {
    int k = i >> 7;
    int n = i & 127;
    sWtP[n * 32 + k] = f2bf(Wp[i]);
  }
  if (tid < HID) {
    sWt1[tid * GINP + (GINP - 1)] = 0;   // zero-pad K = 287
    sWtP[tid * 32 + 31] = 0;             // zero-pad K = 31
    sB1[tid] = b1[tid];
    sBP[tid] = bp[tid];
    sW2[tid] = W2[tid];
  }
  __syncthreads();

  const int wave = tid >> 5;
  const int lane = tid & 31;
  const int tile = blockIdx.x * WAVES_PER_BLOCK + wave;
  const int tileBase = tile * TILE_E;
  if (tileBase >= E) return;

  unsigned short* gbase = &sGate[wave * TILE_E * GINP];

  // ---- gather + stage gate_in = [x[src] | x[dst] | edge_attr] as bf16 rows ----
  for (int e = 0; e < TILE_E; ++e) {
    int edge = tileBase + e;
    int eg = (edge < E) ? edge : (E - 1);
    int s = edge_index[eg];
    int d = edge_index[E + eg];
    const float4* xs = reinterpret_cast<const float4*>(x + (long long)s * NODE_DIM);
    const float4* xd = reinterpret_cast<const float4*>(x + (long long)d * NODE_DIM);
    float4 a = xs[lane];                     // 32 lanes * 4 floats = 128
    float4 b = xd[lane];
    unsigned p0 = (unsigned)f2bf(a.x) | ((unsigned)f2bf(a.y) << 16);
    unsigned p1 = (unsigned)f2bf(a.z) | ((unsigned)f2bf(a.w) << 16);
    *reinterpret_cast<uint2*>(&gbase[e * GINP + lane * 4]) = make_uint2(p0, p1);
    unsigned q0 = (unsigned)f2bf(b.x) | ((unsigned)f2bf(b.y) << 16);
    unsigned q1 = (unsigned)f2bf(b.z) | ((unsigned)f2bf(b.w) << 16);
    *reinterpret_cast<uint2*>(&gbase[e * GINP + NODE_DIM + lane * 4]) = make_uint2(q0, q1);
    if (lane < EDGE_IN)
      gbase[e * GINP + 2 * NODE_DIM + lane] = f2bf(edge_attr[(long long)eg * EDGE_IN + lane]);
    if (lane == 31)
      gbase[e * GINP + GIN] = 0;             // zero-pad col 287
  }
  // Cross-lane LDS store -> load within the wave; drain DS before fragment reads.
  asm volatile("s_wait_dscnt 0" ::: "memory");

  const int Mrow = lane & 15;
  const int hi   = lane >> 4;                // 0 or 1

  // A fragment (16x32 bf16): lane holds M=Mrow, K = kcol + hi*16 + 0..15 (contiguous)
  auto loadA = [&](int kcol) -> v16bf {
    Frag f;
    const unsigned short* p = gbase + Mrow * GINP + kcol + hi * 16;
    f.q[0] = *reinterpret_cast<const uint4*>(p);
    f.q[1] = *reinterpret_cast<const uint4*>(p + 8);
    return f.v;
  };
  // B fragment (32x16 bf16): lane holds N=Mrow, K = ks*32 + hi*16 + 0..15 (contiguous)
  auto loadB1 = [&](int nt, int ks) -> v16bf {
    Frag f;
    const unsigned short* p = &sWt1[(nt * 16 + Mrow) * GINP + ks * 32 + hi * 16];
    f.q[0] = *reinterpret_cast<const uint4*>(p);
    f.q[1] = *reinterpret_cast<const uint4*>(p + 8);
    return f.v;
  };

  // ---- gate MLP layer 1: [16 x 288] @ [288 x 128], 9 K-steps x 8 N-tiles ----
  v8f zero = {};
  v8f acc[8];
#pragma unroll
  for (int nt = 0; nt < 8; ++nt) acc[nt] = zero;

#pragma unroll
  for (int ks = 0; ks < 9; ++ks) {
    v16bf af = loadA(ks * 32);
#pragma unroll
    for (int nt = 0; nt < 8; ++nt) {
      v16bf bf = loadB1(nt, ks);
      acc[nt] = __builtin_amdgcn_wmma_f32_16x16x32_bf16(
          false, af, false, bf, (short)0, acc[nt], false, false);
    }
  }

  // ---- bias + gelu + partial dot with W2 (per-lane over its N columns) ----
  float p[8];
#pragma unroll
  for (int r = 0; r < 8; ++r) p[r] = 0.f;
#pragma unroll
  for (int nt = 0; nt < 8; ++nt) {
    float bv = sB1[nt * 16 + Mrow];
    float wv = sW2[nt * 16 + Mrow];
#pragma unroll
    for (int r = 0; r < 8; ++r) {
      float h = geluf(acc[nt][r] + bv);
      p[r] += h * wv;
    }
  }
  // reduce over the 16 lanes sharing the same M rows (masks < 16 stay in-half)
#pragma unroll
  for (int m = 1; m < 16; m <<= 1) {
#pragma unroll
    for (int r = 0; r < 8; ++r) p[r] += __shfl_xor(p[r], m, 32);
  }
  float b2v = b2[0];
  float gate[8];
#pragma unroll
  for (int r = 0; r < 8; ++r) gate[r] = 1.f / (1.f + expf(-(p[r] + b2v)));

  // ---- projection: edge_attr [16 x 32(pad)] @ Wproj [32 x 128], 8 N-tiles ----
  v8f pacc[8];
#pragma unroll
  for (int nt = 0; nt < 8; ++nt) pacc[nt] = zero;
  {
    v16bf af = loadA(2 * NODE_DIM);          // gate_in cols 256..287 == edge_attr (padded)
#pragma unroll
    for (int nt = 0; nt < 8; ++nt) {
      Frag f;
      const unsigned short* q = &sWtP[(nt * 16 + Mrow) * 32 + hi * 16];
      f.q[0] = *reinterpret_cast<const uint4*>(q);
      f.q[1] = *reinterpret_cast<const uint4*>(q + 8);
      pacc[nt] = __builtin_amdgcn_wmma_f32_16x16x32_bf16(
          false, af, false, f.v, (short)0, pacc[nt], false, false);
    }
  }

  // ---- out = (proj + b_proj) * gate ; lane owns M = r + hi*8, N = nt*16 + Mrow ----
#pragma unroll
  for (int r = 0; r < 8; ++r) {
    int M = r + hi * 8;
    int edge = tileBase + M;
    if (edge >= E) continue;
    float g = gate[r];
    float* orow = out + (long long)edge * EMB_DIM;
#pragma unroll
    for (int nt = 0; nt < 8; ++nt) {
      orow[nt * 16 + Mrow] = (pacc[nt][r] + sBP[nt * 16 + Mrow]) * g;
    }
  }
}

extern "C" void kernel_launch(void* const* d_in, const int* in_sizes, int n_in,
                              void* d_out, int out_size, void* d_ws, size_t ws_size,
                              hipStream_t stream) {
  const float* x         = (const float*)d_in[0];
  const float* edge_attr = (const float*)d_in[1];
  const int*   edge_idx  = (const int*)d_in[2];
  const float* Wp        = (const float*)d_in[3];
  const float* bp        = (const float*)d_in[4];
  const float* W1        = (const float*)d_in[5];
  const float* b1        = (const float*)d_in[6];
  const float* W2        = (const float*)d_in[7];
  const float* b2        = (const float*)d_in[8];
  float* out = (float*)d_out;

  int E = in_sizes[1] / EDGE_IN;                       // 800000
  int tiles  = (E + TILE_E - 1) / TILE_E;              // 50000
  int blocks = (tiles + WAVES_PER_BLOCK - 1) / WAVES_PER_BLOCK;  // 6250
  gated_edge_kernel<<<blocks, BLOCK_THREADS, 0, stream>>>(
      x, edge_attr, edge_idx, Wp, bp, W1, b1, W2, b2, out, E);
  (void)d_ws; (void)ws_size; (void)n_in; (void)out_size;
}
